// PAM_Module_36644660969988
// MI455X (gfx1250) — matrix operand order repaired and checked by reading, not compiled, not run on hardware
//
#include <hip/hip_runtime.h>
#include <hip/hip_bf16.h>

typedef _Float16 h16 __attribute__((ext_vector_type(16)));
typedef _Float16 h8  __attribute__((ext_vector_type(8)));
typedef float    f8  __attribute__((ext_vector_type(8)));
typedef float    f4  __attribute__((ext_vector_type(4)));

#define NPOS 6400
#define CCH  256
#define DQK  128

__device__ __forceinline__ h16 cat8(h8 a, h8 b) {
  return __builtin_shufflevector(a, b, 0,1,2,3,4,5,6,7,8,9,10,11,12,13,14,15);
}
__device__ __forceinline__ h16 ldfrag(const _Float16* p) {
  return cat8(*(const h8*)p, *(const h8*)(p + 16));
}

// 16-lane (half-wave) max reduction in VALU via DPP16.
#define DPP_QUAD_PERM_1032 0xB1
#define DPP_QUAD_PERM_2301 0x4E
#define DPP_ROW_HALF_MIRROR 0x141
#define DPP_ROW_MIRROR 0x140

template <int CTRL>
__device__ __forceinline__ float dpp_bcast(float v) {
  int x = __float_as_int(v);
  int y = __builtin_amdgcn_update_dpp(x, x, CTRL, 0xf, 0xf, false);
  return __int_as_float(y);
}
__device__ __forceinline__ float hmax16(float v) {
  v = fmaxf(v, dpp_bcast<DPP_QUAD_PERM_1032>(v));
  v = fmaxf(v, dpp_bcast<DPP_QUAD_PERM_2301>(v));
  v = fmaxf(v, dpp_bcast<DPP_ROW_HALF_MIRROR>(v));
  v = fmaxf(v, dpp_bcast<DPP_ROW_MIRROR>(v));
  return v;
}

// Issue one async global->LDS 64-byte strip (4 x b128) from one base address.
// INST_OFFSET is added to BOTH the global and LDS addresses, and both layouts
// are contiguous within the strip, so one address setup covers all four.
__device__ __forceinline__ void async_strip64(unsigned lds, const _Float16* gsrc) {
  unsigned long long ga = (unsigned long long)(uintptr_t)gsrc;
  asm volatile("global_load_async_to_lds_b128 %0, %1, off"
               :: "v"(lds), "v"(ga) : "memory");
  asm volatile("global_load_async_to_lds_b128 %0, %1, off offset:16"
               :: "v"(lds), "v"(ga) : "memory");
  asm volatile("global_load_async_to_lds_b128 %0, %1, off offset:32"
               :: "v"(lds), "v"(ga) : "memory");
  asm volatile("global_load_async_to_lds_b128 %0, %1, off offset:48"
               :: "v"(lds), "v"(ga) : "memory");
}

// ---------------------------------------------------------------------------
// Kernel 1: convert x (f32) -> V matrix (f16), same [b][c][n] layout
// ---------------------------------------------------------------------------
__global__ __launch_bounds__(256) void pam_cvt(const float* __restrict__ x,
                                               _Float16* __restrict__ V, int n) {
  int i = blockIdx.x * 256 + threadIdx.x;
  if (i < n) V[i] = (_Float16)x[i];
}

// ---------------------------------------------------------------------------
// Kernel 2: prep — 1x1 convs + BN + relu + sigmoid gate, f32 math,
// writes proj_query (Qm) and proj_key (Km) as f16 [b][n][128] row-major.
// ---------------------------------------------------------------------------
__global__ __launch_bounds__(256) void pam_prep(
    const float* __restrict__ c2, const float* __restrict__ x,
    const float* __restrict__ w_ea1, const float* __restrict__ bn_scale,
    const float* __restrict__ bn_bias, const float* __restrict__ w_ea2,
    const float* __restrict__ b_ea2, const float* __restrict__ w_eq,
    const float* __restrict__ b_eq, const float* __restrict__ w_q,
    const float* __restrict__ b_q,
    _Float16* __restrict__ Qm, _Float16* __restrict__ Km)
{
  __shared__ float tile[256][32];   // 32 KB, reused for x then c2
  __shared__ float qb[64][32];
  __shared__ float eb[64][32];
  __shared__ float hb[64][32];
  __shared__ float eab[32];

  const int b  = blockIdx.y;
  const int n0 = blockIdx.x * 32;
  const int t  = threadIdx.x;

  {
    const float* src = x + ((size_t)b * CCH + t) * NPOS + n0;
    #pragma unroll
    for (int j = 0; j < 8; ++j)
      *(f4*)&tile[t][j * 4] = *(const f4*)(src + j * 4);
  }
  __syncthreads();
  #pragma unroll 1
  for (int i = t; i < 64 * 32; i += 256) {
    int ch = i >> 5, p = i & 31;
    const float* wv = w_q + ch * 256;
    float s = 0.f;
    #pragma unroll 4
    for (int c = 0; c < 256; ++c) s = fmaf(wv[c], tile[c][p], s);
    qb[ch][p] = s + b_q[ch];
  }
  __syncthreads();
  {
    const float* src = c2 + ((size_t)b * CCH + t) * NPOS + n0;
    #pragma unroll
    for (int j = 0; j < 8; ++j)
      *(f4*)&tile[t][j * 4] = *(const f4*)(src + j * 4);
  }
  __syncthreads();
  #pragma unroll 1
  for (int i = t; i < 128 * 32; i += 256) {
    int ch = i >> 5, p = i & 31;
    float s = 0.f;
    if (ch < 64) {
      const float* wv = w_eq + ch * 256;
      #pragma unroll 4
      for (int c = 0; c < 256; ++c) s = fmaf(wv[c], tile[c][p], s);
      eb[ch][p] = s + b_eq[ch];
    } else {
      int k = ch - 64;
      const float* wv = w_ea1 + k * 256;
      #pragma unroll 4
      for (int c = 0; c < 256; ++c) s = fmaf(wv[c], tile[c][p], s);
      float hv = s * (bn_scale[k] * rsqrtf(1.f + 1e-5f)) + bn_bias[k];
      hb[k][p] = fmaxf(hv, 0.f);
    }
  }
  __syncthreads();
  if (t < 32) {
    float s = 0.f;
    #pragma unroll 1
    for (int k = 0; k < 64; ++k) s = fmaf(w_ea2[k], hb[k][t], s);
    eab[t] = 1.f / (1.f + __expf(-(s + b_ea2[0])));
  }
  __syncthreads();
  #pragma unroll 1
  for (int j = t; j < 32 * DQK; j += 256) {
    int p = j >> 7, k = j & 127;
    float base = (k < 64) ? qb[k][p] : eb[k - 64][p];
    float pq   = (k < 64) ? base : eab[p] * base;
    size_t off = ((size_t)b * NPOS + n0 + p) * DQK + k;
    Qm[off] = (_Float16)pq;
    Km[off] = (_Float16)base;
  }
}

// ---------------------------------------------------------------------------
// Kernel 3: fused flash attention, double-buffered async-to-LDS tiles.
// 4 waves/block, 16 query rows per wave, 32 keys per iteration.
// Fragment loads are software-pipelined one step ahead of their WMMA.
// ---------------------------------------------------------------------------
__global__ __launch_bounds__(128) void pam_attn(
    const _Float16* __restrict__ Qm, const _Float16* __restrict__ Km,
    const _Float16* __restrict__ Vm, const float* __restrict__ x,
    const float* __restrict__ gamma, float* __restrict__ out)
{
  __shared__ _Float16 Kt[2][32][136];   // [buf][key][dim], 2 x 8704 B
  __shared__ _Float16 Vt[2][256][40];   // [buf][channel][key], 2 x 20480 B
  __shared__ _Float16 Pt[4][16][32];    // per-wave P staging, 4096 B

  const int b    = blockIdx.y;
  const int m0   = blockIdx.x * 64;
  const int t    = threadIdx.x;
  const int w    = t >> 5;
  const int lane = t & 31;
  const int l16  = lane & 15;
  const bool hi  = lane >= 16;
  const int ho8  = hi ? 8 : 0;
  const int ho16 = hi ? 16 : 0;

  // ---- per-thread async-copy address setup ----
  const int kr = t >> 2, kq = t & 3;
  const _Float16* ksrc = Km + ((size_t)b * NPOS + kr) * DQK + kq * 32;
  const _Float16* vsrc0 = Vm + ((size_t)b * CCH + t) * NPOS;
  const _Float16* vsrc1 = Vm + ((size_t)b * CCH + t + 128) * NPOS;
  unsigned klds[2], vlds0[2], vlds1[2];
  #pragma unroll
  for (int u = 0; u < 2; ++u) {
    klds[u]  = (unsigned)(uintptr_t)&Kt[u][kr][kq * 32];
    vlds0[u] = (unsigned)(uintptr_t)&Vt[u][t][0];
    vlds1[u] = (unsigned)(uintptr_t)&Vt[u][t + 128][0];
  }

  // ---- persistent Q fragments (A-matrix layout, 16x32 f16 per k-chunk) ----
  h16 qf[4];
  {
    const _Float16* qrow = Qm + ((size_t)b * NPOS + m0 + w * 16 + l16) * DQK;
    #pragma unroll
    for (int kc = 0; kc < 4; ++kc)
      qf[kc] = cat8(*(const h8*)(qrow + kc * 32 + ho8),
                    *(const h8*)(qrow + kc * 32 + 16 + ho8));
  }

  // constant all-ones B fragment: D = P x ones -> every column = row sum of P
  h16 onesf;
  #pragma unroll
  for (int i = 0; i < 16; ++i) onesf[i] = (_Float16)1.0f;

  const f8 zero8 = {0.f, 0.f, 0.f, 0.f, 0.f, 0.f, 0.f, 0.f};
  f8 acc[16];
  #pragma unroll
  for (int i = 0; i < 16; ++i) acc[i] = zero8;
  f8 accl = zero8;                 // row-sum accumulator (all columns equal)
  float mrow[8];
  #pragma unroll
  for (int r = 0; r < 8; ++r) mrow[r] = -1e30f;

  // ---- prologue: async-load tile 0 into buffer 0 ----
  async_strip64(klds[0], ksrc);
  async_strip64(vlds0[0], vsrc0);
  async_strip64(vlds1[0], vsrc1);

  const int NITER = NPOS / 32;
  #pragma unroll 1
  for (int it = 0; it < NITER; ++it) {
    const int cur = it & 1;
    // my previous-tile async copies must be in LDS before anyone reads them
    asm volatile("s_wait_asynccnt 0" ::: "memory");
    __syncthreads();
    // prefetch next tile into the other buffer; its last readers finished
    // before the barrier above, so this cannot race.
    if (it + 1 < NITER) {
      const int n1 = (it + 1) * 32;
      async_strip64(klds[cur ^ 1], ksrc + (size_t)n1 * DQK);
      async_strip64(vlds0[cur ^ 1], vsrc0 + n1);
      async_strip64(vlds1[cur ^ 1], vsrc1 + n1);
    }

    // ---- S = Q * K^T : two 16x16 tiles, k = 128, pipelined frag loads ----
    const _Float16* kbase0 = &Kt[cur][l16][ho8];
    const _Float16* kbase1 = &Kt[cur][16 + l16][ho8];
    h16 b0 = ldfrag(kbase0);
    h16 b1 = ldfrag(kbase1);
    f8 s0 = zero8, s1 = zero8;
    #pragma unroll
    for (int kc = 0; kc < 4; ++kc) {
      h16 n0f{}, n1f{};
      if (kc < 3) {
        n0f = ldfrag(kbase0 + (kc + 1) * 32);
        n1f = ldfrag(kbase1 + (kc + 1) * 32);
      }
      s0 = __builtin_amdgcn_wmma_f32_16x16x32_f16(false, qf[kc], false, b0,
                                                  (short)0, s0, false, false);
      s1 = __builtin_amdgcn_wmma_f32_16x16x32_f16(false, qf[kc], false, b1,
                                                  (short)0, s1, false, false);
      if (kc < 3) { b0 = n0f; b1 = n1f; }
    }

    // ---- online softmax (wave-private, DPP16 max reductions) ----
    f8 av;
    #pragma unroll
    for (int r = 0; r < 8; ++r) {
      float v0 = s0[r], v1 = s1[r];
      float rm = hmax16(fmaxf(v0, v1));
      float mn = fmaxf(mrow[r], rm);
      av[r] = __expf(mrow[r] - mn);
      mrow[r] = mn;
      float p0 = __expf(v0 - mn);
      float p1 = __expf(v1 - mn);
      int prow = r + ho8;
      Pt[w][prow][l16]      = (_Float16)p0;
      Pt[w][prow][16 + l16] = (_Float16)p1;
    }
    // vector rescale of all accumulators by alpha; also covers P bounce latency
    #pragma unroll
    for (int c16 = 0; c16 < 16; ++c16) acc[c16] *= av;
    accl *= av;

    // ---- P fragment (A layout, 16x32) from per-wave LDS bounce ----
    const _Float16* pp = &Pt[w][l16][0];
    h16 pf = cat8(*(const h8*)(pp + ho8), *(const h8*)(pp + 16 + ho8));

    // ---- O += P * V over 256 channels (16 tiles) + row-sum tile ----
    const _Float16* vbase = &Vt[cur][l16][ho8];
    h16 vf = ldfrag(vbase);
    #pragma unroll
    for (int c16 = 0; c16 < 16; ++c16) {
      h16 nvf{};
      if (c16 < 15) nvf = ldfrag(vbase + (c16 + 1) * 16 * 40);
      acc[c16] = __builtin_amdgcn_wmma_f32_16x16x32_f16(false, pf, false, vf,
                                                        (short)0, acc[c16],
                                                        false, false);
      if (c16 < 15) vf = nvf;
    }
    accl = __builtin_amdgcn_wmma_f32_16x16x32_f16(false, pf, false, onesf,
                                                  (short)0, accl, false, false);
  }

  // ---- epilogue: out = gamma * (O / l) + x ----
  const float g = gamma[0];
  float rinv[8];
  #pragma unroll
  for (int r = 0; r < 8; ++r) rinv[r] = 1.0f / accl[r];
  #pragma unroll 1
  for (int c16 = 0; c16 < 16; ++c16) {
    const int ch = c16 * 16 + l16;
    #pragma unroll
    for (int r = 0; r < 8; ++r) {
      const int m = m0 + w * 16 + r + ho8;
      const size_t idx = ((size_t)b * CCH + ch) * NPOS + m;
      out[idx] = g * (acc[c16][r] * rinv[r]) + x[idx];
    }
  }
}

// ---------------------------------------------------------------------------
extern "C" void kernel_launch(void* const* d_in, const int* in_sizes, int n_in,
                              void* d_out, int out_size, void* d_ws, size_t ws_size,
                              hipStream_t stream) {
  (void)in_sizes; (void)n_in; (void)out_size; (void)ws_size;

  const float* c2       = (const float*)d_in[0];
  const float* x        = (const float*)d_in[1];
  const float* w_ea1    = (const float*)d_in[2];
  const float* bn_scale = (const float*)d_in[3];
  const float* bn_bias  = (const float*)d_in[4];
  const float* w_ea2    = (const float*)d_in[5];
  const float* b_ea2    = (const float*)d_in[6];
  const float* w_eq     = (const float*)d_in[7];
  const float* b_eq     = (const float*)d_in[8];
  const float* w_q      = (const float*)d_in[9];
  const float* b_q      = (const float*)d_in[10];
  const float* gamma    = (const float*)d_in[11];
  float* out = (float*)d_out;

  _Float16* Qm = (_Float16*)d_ws;                       // 2*6400*128 f16
  _Float16* Km = Qm + (size_t)2 * NPOS * DQK;           // 2*6400*128 f16
  _Float16* Vm = Km + (size_t)2 * NPOS * DQK;           // 2*256*6400 f16

  const int nv = 2 * CCH * NPOS;
  pam_cvt<<<(nv + 255) / 256, 256, 0, stream>>>(x, Vm, nv);
  pam_prep<<<dim3(NPOS / 32, 2), 256, 0, stream>>>(
      c2, x, w_ea1, bn_scale, bn_bias, w_ea2, b_ea2, w_eq, b_eq, w_q, b_q,
      Qm, Km);
  pam_attn<<<dim3(NPOS / 64, 2), 128, 0, stream>>>(Qm, Km, Vm, x, gamma, out);
}
